// DiTBlockTriton_77421080477791
// MI455X (gfx1250) — 16-bit and 8-bit matrix operand order repaired and checked by reading, compiled, not run on hardware
//
#include <hip/hip_runtime.h>
#include <hip/hip_bf16.h>

typedef __attribute__((ext_vector_type(8)))  int      v8i;
typedef __attribute__((ext_vector_type(8)))  float    v8f;
typedef __attribute__((ext_vector_type(16))) _Float16 v16h;
typedef __attribute__((ext_vector_type(8)))  _Float16 h8;

#define B_     8
#define N_     1024
#define DIM_   1024
#define MTOK_  (B_ * N_)
#define HID_   4096
#define HEADS_ 16
#define HD_    64
#define QKVN_  (3 * DIM_)


__device__ __forceinline__ float waveRedSum(float v) {
#pragma unroll
  for (int m = 16; m > 0; m >>= 1) v += __shfl_xor(v, m, 32);
  return v;
}
__device__ __forceinline__ float waveRedMax(float v) {
#pragma unroll
  for (int m = 16; m > 0; m >>= 1) v = fmaxf(v, __shfl_xor(v, m, 32));
  return v;
}
__device__ __forceinline__ float gelu1(float x) {
  return 0.5f * x * (1.0f + erff(x * 0.70710678118654752f));
}


__global__ void __launch_bounds__(256) k_absmean(const float* __restrict__ W, int n,
                                                 float* __restrict__ accum) {
  float s = 0.f;
  for (int i = blockIdx.x * 256 + threadIdx.x; i < n; i += gridDim.x * 256)
    s += fabsf(W[i]);
  s = waveRedSum(s);
  __shared__ float red[8];
  int lane = threadIdx.x & 31, wave = threadIdx.x >> 5;
  if (lane == 0) red[wave] = s;
  __syncthreads();
  if (threadIdx.x == 0) {
    float t = 0.f;
    for (int i = 0; i < 8; i++) t += red[i];
    atomicAdd(accum, t);
  }
}

__global__ void k_wfinal(const float* __restrict__ s, float* __restrict__ d) {
  const float cnt[4] = {3145728.f, 1048576.f, 4194304.f, 4194304.f};
  int i = threadIdx.x;
  if (i < 4) d[i] = fmaxf(s[i] / cnt[i], 1e-5f);
}

__global__ void __launch_bounds__(256) k_quant_w(const float* __restrict__ W,
                                                 signed char* __restrict__ Q, int n4,
                                                 const float* __restrict__ wdq) {
  int i = blockIdx.x * 256 + threadIdx.x;
  if (i >= n4) return;
  float inv = 1.f / wdq[0];
  float4 v = ((const float4*)W)[i];
  char4 q;
  q.x = (signed char)fminf(1.f, fmaxf(-1.f, roundf(v.x * inv)));
  q.y = (signed char)fminf(1.f, fmaxf(-1.f, roundf(v.y * inv)));
  q.z = (signed char)fminf(1.f, fmaxf(-1.f, roundf(v.z * inv)));
  q.w = (signed char)fminf(1.f, fmaxf(-1.f, roundf(v.w * inv)));
  ((char4*)Q)[i] = q;
}


__global__ void __launch_bounds__(256) k_modulation(const float* __restrict__ T,
                                                    const float* __restrict__ W,
                                                    const float* __restrict__ bias,
                                                    float* __restrict__ out) {
  int b = blockIdx.y;
  int j = blockIdx.x * 256 + threadIdx.x;
  __shared__ float st[DIM_];
  for (int i = threadIdx.x; i < DIM_; i += 256) {
    float v = T[b * DIM_ + i];
    st[i] = v / (1.f + __expf(-v));
  }
  __syncthreads();
  const float4* w4 = (const float4*)(W + (size_t)j * DIM_);
  const float4* s4 = (const float4*)st;
  float acc = 0.f;
  for (int d = 0; d < DIM_ / 4; d++) {
    float4 wv = w4[d], sv = s4[d];
    acc += wv.x * sv.x + wv.y * sv.y + wv.z * sv.z + wv.w * sv.w;
  }
  out[(size_t)b * (4 * DIM_) + j] = acc + bias[j];
}


__global__ void __launch_bounds__(256) k_rmsnorm_mod_quant(
    const float* __restrict__ X, const float* __restrict__ g,
    const float* __restrict__ mod, int shiftCol, int scaleCol,
    signed char* __restrict__ Q, float* __restrict__ adq) {
  const int row = blockIdx.x;
  const int b = row >> 10;
  const int tid = threadIdx.x, lane = tid & 31, wave = tid >> 5;
  const float* xr = X + (size_t)row * DIM_;
  const float* sh = mod + (size_t)b * (4 * DIM_) + shiftCol;
  const float* sc = mod + (size_t)b * (4 * DIM_) + scaleCol;
  __shared__ float red1[8], red2[8];

  float4 xv = ((const float4*)xr)[tid];
  float ss = xv.x * xv.x + xv.y * xv.y + xv.z * xv.z + xv.w * xv.w;
  ss = waveRedSum(ss);
  if (lane == 0) red1[wave] = ss;
  __syncthreads();
  float tot = 0.f;
  for (int i = 0; i < 8; i++) tot += red1[i];
  float rstd = rsqrtf(tot * (1.0f / (float)DIM_) + 1e-6f);

  float4 gv = ((const float4*)g)[tid];
  float4 shv = ((const float4*)sh)[tid];
  float4 scv = ((const float4*)sc)[tid];
  float h0 = xv.x * rstd * gv.x * (1.f + scv.x) + shv.x;
  float h1 = xv.y * rstd * gv.y * (1.f + scv.y) + shv.y;
  float h2 = xv.z * rstd * gv.z * (1.f + scv.z) + shv.z;
  float h3 = xv.w * rstd * gv.w * (1.f + scv.w) + shv.w;

  float amax = fmaxf(fmaxf(fabsf(h0), fabsf(h1)), fmaxf(fabsf(h2), fabsf(h3)));
  amax = waveRedMax(amax);
  if (lane == 0) red2[wave] = amax;
  __syncthreads();
  float am = red2[0];
  for (int i = 1; i < 8; i++) am = fmaxf(am, red2[i]);
  float dq = fmaxf(am, 1e-5f) * (1.f / 127.f);
  float s = 1.f / dq;

  char4 q;
  q.x = (signed char)fminf(127.f, fmaxf(-128.f, roundf(h0 * s)));
  q.y = (signed char)fminf(127.f, fmaxf(-128.f, roundf(h1 * s)));
  q.z = (signed char)fminf(127.f, fmaxf(-128.f, roundf(h2 * s)));
  q.w = (signed char)fminf(127.f, fmaxf(-128.f, roundf(h3 * s)));
  ((char4*)(Q + (size_t)row * DIM_))[tid] = q;
  if (tid == 0) adq[row] = dq;
}


__global__ void __launch_bounds__(256) k_act_quant(const float* __restrict__ X,
                                                   signed char* __restrict__ Q,
                                                   float* __restrict__ adq, int cols,
                                                   int dogelu) {
  const int row = blockIdx.x;
  const int tid = threadIdx.x, lane = tid & 31, wave = tid >> 5;
  const float* xr = X + (size_t)row * cols;
  const int n4 = cols >> 2;
  __shared__ float red[8];
  float amax = 0.f;
  for (int i = tid; i < n4; i += 256) {
    float4 v = ((const float4*)xr)[i];
    if (dogelu) { v.x = gelu1(v.x); v.y = gelu1(v.y); v.z = gelu1(v.z); v.w = gelu1(v.w); }
    amax = fmaxf(amax, fmaxf(fmaxf(fabsf(v.x), fabsf(v.y)), fmaxf(fabsf(v.z), fabsf(v.w))));
  }
  amax = waveRedMax(amax);
  if (lane == 0) red[wave] = amax;
  __syncthreads();
  float am = red[0];
  for (int i = 1; i < 8; i++) am = fmaxf(am, red[i]);
  float dq = fmaxf(am, 1e-5f) * (1.f / 127.f);
  float s = 1.f / dq;
  for (int i = tid; i < n4; i += 256) {
    float4 v = ((const float4*)xr)[i];
    if (dogelu) { v.x = gelu1(v.x); v.y = gelu1(v.y); v.z = gelu1(v.z); v.w = gelu1(v.w); }
    char4 q;
    q.x = (signed char)fminf(127.f, fmaxf(-128.f, roundf(v.x * s)));
    q.y = (signed char)fminf(127.f, fmaxf(-128.f, roundf(v.y * s)));
    q.z = (signed char)fminf(127.f, fmaxf(-128.f, roundf(v.z * s)));
    q.w = (signed char)fminf(127.f, fmaxf(-128.f, roundf(v.w * s)));
    ((char4*)(Q + (size_t)row * cols))[i] = q;
  }
  if (tid == 0) adq[row] = dq;
}


__global__ void __launch_bounds__(256) k_gemm_iu8(
    const signed char* __restrict__ A, int lda,
    const signed char* __restrict__ Bw, int ldb,
    float* __restrict__ C, int ldc,
    const float* __restrict__ ascl, const float* __restrict__ wdq,
    const float* __restrict__ res, int addres, int K) {
  const int lane = threadIdx.x & 31;
  const int wave = threadIdx.x >> 5;
  const int laneHi = lane >> 4;
  const int ln = lane & 15;
  const int m0 = blockIdx.y * 128 + (wave & 3) * 32;
  const int n0 = blockIdx.x * 64 + (wave >> 2) * 32;
  const float wsc = wdq[0];

  v8i zero = {0, 0, 0, 0, 0, 0, 0, 0};
  v8i acc[2][2] = {{zero, zero}, {zero, zero}};

  for (int ks = 0; ks < K; ks += 64) {
    v8i af[2], bf[2];
#pragma unroll
    for (int i = 0; i < 2; i++) {
      const signed char* rp = A + (size_t)(m0 + 16 * i + ln) * lda + ks + (laneHi ? 8 : 0);
      int2 c0 = *(const int2*)(rp);
      int2 c1 = *(const int2*)(rp + 16);
      int2 c2 = *(const int2*)(rp + 32);
      int2 c3 = *(const int2*)(rp + 48);
      v8i f;
      f[0] = c0.x; f[1] = c0.y; f[2] = c1.x; f[3] = c1.y;
      f[4] = c2.x; f[5] = c2.y; f[6] = c3.x; f[7] = c3.y;
      af[i] = f;
    }
#pragma unroll
    for (int j = 0; j < 2; j++) {
      const signed char* cp = Bw + (size_t)(n0 + 16 * j + ln) * ldb + ks + (laneHi ? 8 : 0);
      int2 d0 = *(const int2*)(cp);
      int2 d1 = *(const int2*)(cp + 16);
      int2 d2 = *(const int2*)(cp + 32);
      int2 d3 = *(const int2*)(cp + 48);
      v8i f;
      f[0] = d0.x; f[1] = d0.y; f[2] = d1.x; f[3] = d1.y;
      f[4] = d2.x; f[5] = d2.y; f[6] = d3.x; f[7] = d3.y;
      bf[j] = f;
    }
#pragma unroll
    for (int i = 0; i < 2; i++)
#pragma unroll
      for (int j = 0; j < 2; j++)
        acc[i][j] = __builtin_amdgcn_wmma_i32_16x16x64_iu8(
            true, af[i], true, bf[j], acc[i][j], false, false);
  }

#pragma unroll
  for (int i = 0; i < 2; i++)
#pragma unroll
    for (int j = 0; j < 2; j++) {
      v8i a = acc[i][j];
#pragma unroll
      for (int r = 0; r < 8; r++) {
        int m = m0 + 16 * i + r + 8 * laneHi;
        int n = n0 + 16 * j + ln;
        float v = (float)a[r] * (ascl[m] * wsc);
        if (addres) v += res[(size_t)m * ldc + n];
        C[(size_t)m * ldc + n] = v;
      }
    }
}


__global__ void __launch_bounds__(128) k_attention(const float* __restrict__ qkv,
                                                   float* __restrict__ O) {
  const int LD = QKVN_;
  const int b = blockIdx.z, h = blockIdx.y;
  const int wave = threadIdx.x >> 5, lane = threadIdx.x & 31;
  const int ln = lane & 15, laneHi = lane >> 4;
  const int q0 = blockIdx.x * 64 + wave * 16;

  __shared__ _Float16 Ks[32][HD_];
  __shared__ _Float16 Vt[HD_][32];
  __shared__ _Float16 Pl[4][16][32];

  v16h aq[2];
  {
    const float* qr = qkv + (size_t)(b * N_ + q0 + ln) * LD + h * HD_;
#pragma unroll
    for (int c = 0; c < 2; c++) {
      int kb = 32 * c + (laneHi ? 8 : 0);
#pragma unroll
      for (int u = 0; u < 8; u++) aq[c][u] = (_Float16)qr[kb + u];
#pragma unroll
      for (int u = 0; u < 8; u++) aq[c][8 + u] = (_Float16)qr[kb + 16 + u];
    }
  }

  float mrow[8], lrow[8];
  v8f zf = {0, 0, 0, 0, 0, 0, 0, 0};
  v8f of[4] = {zf, zf, zf, zf};
#pragma unroll
  for (int r = 0; r < 8; r++) { mrow[r] = -1e30f; lrow[r] = 0.f; }

  for (int kt = 0; kt < N_; kt += 32) {
    for (int idx = threadIdx.x; idx < 32 * HD_; idx += 128) {
      int kk = idx >> 6, f = idx & 63;
      const float* base = qkv + (size_t)(b * N_ + kt + kk) * LD + h * HD_;
      Ks[kk][f] = (_Float16)base[DIM_ + f];
      Vt[f][kk] = (_Float16)base[2 * DIM_ + f];
    }
    __syncthreads();

    v8f s[2];
#pragma unroll
    for (int j = 0; j < 2; j++) {
      v8f a = zf;
#pragma unroll
      for (int c = 0; c < 2; c++) {
        v16h bk;
        int key = 16 * j + ln;
        int fb = 32 * c + (laneHi ? 8 : 0);
        h8 p0 = *(const h8*)&Ks[key][fb];
        h8 p1 = *(const h8*)&Ks[key][fb + 16];
#pragma unroll
        for (int u = 0; u < 8; u++) { bk[u] = p0[u]; bk[8 + u] = p1[u]; }
        a = __builtin_amdgcn_wmma_f32_16x16x32_f16(false, aq[c], false, bk,
                                                   (short)0, a, false, false);
      }
      s[j] = a;
    }

#pragma unroll
    for (int r = 0; r < 8; r++) {
      float s0 = s[0][r] * 0.125f, s1 = s[1][r] * 0.125f;
      float mx = fmaxf(s0, s1);
#pragma unroll
      for (int msk = 8; msk > 0; msk >>= 1) mx = fmaxf(mx, __shfl_xor(mx, msk, 32));
      float mn = fmaxf(mrow[r], mx);
      float al = __expf(mrow[r] - mn);
      s0 = __expf(s0 - mn);
      s1 = __expf(s1 - mn);
      float rs = s0 + s1;
#pragma unroll
      for (int msk = 8; msk > 0; msk >>= 1) rs += __shfl_xor(rs, msk, 32);
      lrow[r] = lrow[r] * al + rs;
      mrow[r] = mn;
#pragma unroll
      for (int nt = 0; nt < 4; nt++) of[nt][r] *= al;
      Pl[wave][r + 8 * laneHi][ln] = (_Float16)s0;
      Pl[wave][r + 8 * laneHi][16 + ln] = (_Float16)s1;
    }

    asm volatile("s_wait_dscnt 0x0" ::: "memory");

    v16h ap;
    {
      int kb = laneHi ? 8 : 0;
      h8 p0 = *(const h8*)&Pl[wave][ln][kb];
      h8 p1 = *(const h8*)&Pl[wave][ln][kb + 16];
#pragma unroll
      for (int u = 0; u < 8; u++) { ap[u] = p0[u]; ap[8 + u] = p1[u]; }
    }

#pragma unroll
    for (int nt = 0; nt < 4; nt++) {
      v16h bv;
      int f = 16 * nt + ln;
      int kb2 = laneHi ? 8 : 0;
      h8 r0 = *(const h8*)&Vt[f][kb2];
      h8 r1 = *(const h8*)&Vt[f][kb2 + 16];
#pragma unroll
      for (int u = 0; u < 8; u++) { bv[u] = r0[u]; bv[8 + u] = r1[u]; }
      of[nt] = __builtin_amdgcn_wmma_f32_16x16x32_f16(false, ap, false, bv,
                                                      (short)0, of[nt], false, false);
    }
    __syncthreads();
  }

#pragma unroll
  for (int nt = 0; nt < 4; nt++)
#pragma unroll
    for (int r = 0; r < 8; r++) {
      int row = q0 + r + 8 * laneHi;
      int col = h * HD_ + 16 * nt + ln;
      O[(size_t)(b * N_ + row) * DIM_ + col] = of[nt][r] / lrow[r];
    }
}


extern "C" void kernel_launch(void* const* d_in, const int* in_sizes, int n_in,
                              void* d_out, int out_size, void* d_ws, size_t ws_size,
                              hipStream_t stream) {
  (void)in_sizes; (void)n_in; (void)out_size; (void)ws_size;
  const float* x       = (const float*)d_in[0];
  const float* t_emb   = (const float*)d_in[1];
  const float* w_qkv   = (const float*)d_in[2];
  const float* w_proj  = (const float*)d_in[3];
  const float* w_fc1   = (const float*)d_in[4];
  const float* w_fc2   = (const float*)d_in[5];
  const float* w_adaln = (const float*)d_in[6];
  const float* b_adaln = (const float*)d_in[7];
  const float* g1      = (const float*)d_in[8];
  const float* g2      = (const float*)d_in[9];
  float* out = (float*)d_out;

  char* p = (char*)d_ws;
  float* mod = (float*)p;                 p += (size_t)B_ * 4 * DIM_ * 4;
  float* wsum = (float*)p;                p += 256;
  float* wdq = (float*)p;                 p += 256;
  signed char* wq_qkv = (signed char*)p;  p += (size_t)QKVN_ * DIM_;
  signed char* wq_proj = (signed char*)p; p += (size_t)DIM_ * DIM_;
  signed char* wq_fc1 = (signed char*)p;  p += (size_t)HID_ * DIM_;
  signed char* wq_fc2 = (signed char*)p;  p += (size_t)DIM_ * HID_;
  signed char* hq = (signed char*)p;      p += (size_t)MTOK_ * HID_;
  float* arow = (float*)p;                p += (size_t)MTOK_ * 4;
  float* big = (float*)p;                 p += (size_t)MTOK_ * HID_ * 4;
  float* obuf = (float*)p;                p += (size_t)MTOK_ * DIM_ * 4;
  float* x1 = (float*)p;                  p += (size_t)MTOK_ * DIM_ * 4;

  hipMemsetAsync(wsum, 0, 16, stream);
  k_absmean<<<512, 256, 0, stream>>>(w_qkv, QKVN_ * DIM_, wsum + 0);
  k_absmean<<<512, 256, 0, stream>>>(w_proj, DIM_ * DIM_, wsum + 1);
  k_absmean<<<512, 256, 0, stream>>>(w_fc1, HID_ * DIM_, wsum + 2);
  k_absmean<<<512, 256, 0, stream>>>(w_fc2, DIM_ * HID_, wsum + 3);
  k_wfinal<<<1, 4, 0, stream>>>(wsum, wdq);
  k_quant_w<<<(QKVN_ * DIM_ / 4 + 255) / 256, 256, 0, stream>>>(w_qkv, wq_qkv, QKVN_ * DIM_ / 4, wdq + 0);
  k_quant_w<<<(DIM_ * DIM_ / 4 + 255) / 256, 256, 0, stream>>>(w_proj, wq_proj, DIM_ * DIM_ / 4, wdq + 1);
  k_quant_w<<<(HID_ * DIM_ / 4 + 255) / 256, 256, 0, stream>>>(w_fc1, wq_fc1, HID_ * DIM_ / 4, wdq + 2);
  k_quant_w<<<(DIM_ * HID_ / 4 + 255) / 256, 256, 0, stream>>>(w_fc2, wq_fc2, DIM_ * HID_ / 4, wdq + 3);

  k_modulation<<<dim3(4 * DIM_ / 256, B_), 256, 0, stream>>>(t_emb, w_adaln, b_adaln, mod);

  k_rmsnorm_mod_quant<<<MTOK_, 256, 0, stream>>>(x, g1, mod, 0, DIM_, hq, arow);
  k_gemm_iu8<<<dim3(QKVN_ / 64, MTOK_ / 128), 256, 0, stream>>>(
      hq, DIM_, wq_qkv, DIM_, big, QKVN_, arow, wdq + 0, nullptr, 0, DIM_);
  k_attention<<<dim3(N_ / 64, HEADS_, B_), 128, 0, stream>>>(big, obuf);
  k_act_quant<<<MTOK_, 256, 0, stream>>>(obuf, hq, arow, DIM_, 0);
  k_gemm_iu8<<<dim3(DIM_ / 64, MTOK_ / 128), 256, 0, stream>>>(
      hq, DIM_, wq_proj, DIM_, x1, DIM_, arow, wdq + 1, x, 1, DIM_);

  k_rmsnorm_mod_quant<<<MTOK_, 256, 0, stream>>>(x1, g2, mod, 2 * DIM_, 3 * DIM_, hq, arow);
  k_gemm_iu8<<<dim3(HID_ / 64, MTOK_ / 128), 256, 0, stream>>>(
      hq, DIM_, wq_fc1, DIM_, big, HID_, arow, wdq + 2, nullptr, 0, DIM_);
  k_act_quant<<<MTOK_, 256, 0, stream>>>(big, hq, arow, HID_, 1);
  k_gemm_iu8<<<dim3(DIM_ / 64, MTOK_ / 128), 256, 0, stream>>>(
      hq, HID_, wq_fc2, HID_, out, DIM_, arow, wdq + 3, x1, 1, HID_);
}
